// TinyRecursiveModelTRMv6_63754494542343
// MI455X (gfx1250) — compile-verified, matrix operand order checked by reading, not run on hardware
//
#include <hip/hip_runtime.h>
#include <hip/hip_bf16.h>
#include <math.h>

// ---------------------------------------------------------------------------
// TinyRecursiveModel TRMv6 on MI455X (gfx1250, wave32, WMMA 16x16x32 f16/f32)
// One wave (32 lanes) owns a 16-row tile (M=16 == WMMA M).
// GEMM micro-kernel: A fragment from LDS (2x ds_load_b128) reused across a
// group of 4 N-tiles. __launch_bounds__(32,4) caps waves at <=256 VGPRs so the
// scheduler cannot pipeline past the VGPR-MSB window (no s_set_vgpr_msb).
// ---------------------------------------------------------------------------

typedef __attribute__((ext_vector_type(16))) _Float16 v16h;
typedef __attribute__((ext_vector_type(8)))  float    v8f;

#define DEV __device__ __forceinline__

constexpr int    BATCH = 65536;
constexpr int    DIM   = 128;
constexpr size_t BD    = (size_t)BATCH * DIM;
constexpr int    KOUT  = 5;
constexpr float  EPS   = 1e-5f;

DEV float gelu_erf(float v) { return 0.5f * v * (1.0f + erff(v * 0.70710678118654752f)); }

// combine the two lanes (l, l^16) that co-own one row
DEV float pair_red(float v) { return v + __shfl_xor(v, 16, 32); }

// ---------------------------------------------------------------------------
// Register-blocked WMMA micro-kernel: 16 rows x (NGROUP*16) cols, K = KTILES*32.
// A (f16) in LDS row-major [16][ldk]; per kt the lane's 16 halves are two
// contiguous 16B chunks at (m*ldk + kt*32 + g*8) and (+16) -> 2x ds_load_b128.
// B pre-packed per (kt,nt) tile: 512 halves, lane-major, one 32B load per lane.
// ---------------------------------------------------------------------------
template <int KTILES, int NGROUP>
DEV void gemm_block(const _Float16* __restrict__ aLds, int ldk,
                    const _Float16* __restrict__ Bp, int nt0, int ntiles,
                    int lane, v8f* accs) {
    const int m = lane & 15, g = lane >> 4;
    const _Float16* __restrict__ arow = aLds + m * ldk + g * 8;
    const _Float16* __restrict__ bbase = Bp + ((size_t)nt0 << 9) + lane * 16;
#pragma unroll
    for (int kt = 0; kt < KTILES; ++kt) {
        v16h a;
#pragma unroll
        for (int h = 0; h < 8; ++h) {
            a[h]     = arow[kt * 32 + h];
            a[8 + h] = arow[kt * 32 + 16 + h];
        }
#pragma unroll
        for (int q = 0; q < NGROUP; ++q) {
            const v16h b = *(const v16h*)(bbase + (((size_t)(kt * ntiles + q)) << 9));
            accs[q] = __builtin_amdgcn_wmma_f32_16x16x32_f16(false, a, false, b,
                                                             (short)0, accs[q], false, false);
        }
    }
}

DEV void zero_accs4(v8f* a) {
#pragma unroll
    for (int q = 0; q < 4; ++q) a[q] = (v8f){0.f, 0.f, 0.f, 0.f, 0.f, 0.f, 0.f, 0.f};
}

// ---------------------------------------------------------------------------
// Weight packer: W[K][N] f32 -> tile(kt,nt) of 512 halves, lane-major, mirroring
// the B 32x16 fragment layout (N = lane&15; K pattern matches the A layout).
// ---------------------------------------------------------------------------
__global__ void pack_b_kernel(const float* __restrict__ W, _Float16* __restrict__ P,
                              int Kd, int Nd) {
    const int idx = blockIdx.x * blockDim.x + threadIdx.x;
    if (idx >= Kd * Nd) return;
    const int tile = idx >> 9, rem = idx & 511;
    const int lane = rem >> 4, h = rem & 15;
    const int ntiles = Nd >> 4;
    const int kt = tile / ntiles, nt = tile - kt * ntiles;
    const int v = h >> 1, hh = h & 1, g = lane >> 4;
    const int k = kt * 32 + (v & 3) * 2 + hh + g * 8 + (v >> 2) * 16;
    const int n = nt * 16 + (lane & 15);
    P[idx] = (_Float16)W[k * Nd + n];
}

// ---------------------------------------------------------------------------
__global__ void kinit(float* __restrict__ zbuf, float* __restrict__ ybuf,
                      const float* __restrict__ y_init, float* __restrict__ scal) {
    size_t idx = (size_t)blockIdx.x * blockDim.x + threadIdx.x;
    const size_t stride = (size_t)gridDim.x * blockDim.x;
    for (; idx < BD; idx += stride) { zbuf[idx] = 0.f; ybuf[idx] = y_init[idx]; }
    if (blockIdx.x == 0 && threadIdx.x < 16) scal[threadIdx.x] = 0.f;
}

// ---------------------------------------------------------------------------
// K1: 4 inner latent-MLP steps + affect/gate, write z and history snapshot.
// ---------------------------------------------------------------------------
__global__ __launch_bounds__(32, 4) void k1_inner(
    const float* __restrict__ x, const float* __restrict__ ybuf, float* __restrict__ zbuf,
    _Float16* __restrict__ hist_k,
    const float* __restrict__ lg, const float* __restrict__ lb,
    const float* __restrict__ zg, const float* __restrict__ zb,
    const _Float16* __restrict__ w1p, const float* __restrict__ b1,
    const _Float16* __restrict__ w2p, const float* __restrict__ b2,
    const float* __restrict__ affect_vec,
    const float* __restrict__ aff_w, const float* __restrict__ aff_b,
    const float* __restrict__ gate_w, const float* __restrict__ gate_b,
    float* __restrict__ conf_sum) {
    __shared__ float    xsh[16 * 128], ysh[16 * 128], zsh[16 * 128];
    __shared__ _Float16 a16[16 * 384];
    __shared__ _Float16 h16[16 * 256];

    const int    l = threadIdx.x, m = l & 15, g = l >> 4;
    const size_t row0 = (size_t)blockIdx.x * 16;
    const size_t base = row0 * 128;

    for (int i = 0; i < 64; ++i) {
        const int e = l + 32 * i;
        xsh[e] = x[base + e]; ysh[e] = ybuf[base + e]; zsh[e] = zbuf[base + e];
    }
    __syncthreads();

    for (int inner = 0; inner < 4; ++inner) {
        // ---- LayerNorm over concat(x,y,z) [384] -> a16 (f16) ----
        float s = 0.f, sq = 0.f;
        for (int i = 0; i < 192; ++i) {
            const int c = g * 192 + i;
            const float v = (c < 128) ? xsh[m * 128 + c]
                          : (c < 256) ? ysh[m * 128 + c - 128]
                                      : zsh[m * 128 + c - 256];
            s += v; sq += v * v;
        }
        s = pair_red(s); sq = pair_red(sq);
        const float mean = s * (1.f / 384.f);
        const float rstd = rsqrtf(sq * (1.f / 384.f) - mean * mean + EPS);
        for (int i = 0; i < 192; ++i) {
            const int c = g * 192 + i;
            const float v = (c < 128) ? xsh[m * 128 + c]
                          : (c < 256) ? ysh[m * 128 + c - 128]
                                      : zsh[m * 128 + c - 256];
            a16[m * 384 + c] = (_Float16)((v - mean) * rstd * lg[c] + lb[c]);
        }
        __syncthreads();

        // ---- GEMM1 [16x384]x[384x256] + GELU ----
#pragma unroll 1
        for (int nt0 = 0; nt0 < 16; nt0 += 4) {
            v8f accs[4];
            zero_accs4(accs);
            gemm_block<12, 4>(a16, 384, w1p, nt0, 16, l, accs);
#pragma unroll
            for (int q = 0; q < 4; ++q)
#pragma unroll
                for (int j = 0; j < 8; ++j) {
                    const int mm = j + 8 * g, n = (nt0 + q) * 16 + m;
                    h16[mm * 256 + n] = (_Float16)gelu_erf(accs[q][j] + b1[n]);
                }
        }
        __syncthreads();

        // ---- GEMM2 [16x256]x[256x128], z += 0.3*delta ----
#pragma unroll 1
        for (int nt0 = 0; nt0 < 8; nt0 += 4) {
            v8f accs[4];
            zero_accs4(accs);
            gemm_block<8, 4>(h16, 256, w2p, nt0, 8, l, accs);
#pragma unroll
            for (int q = 0; q < 4; ++q)
#pragma unroll
                for (int j = 0; j < 8; ++j) {
                    const int mm = j + 8 * g, n = (nt0 + q) * 16 + m;
                    zsh[mm * 128 + n] += 0.3f * (accs[q][j] + b2[n]);
                }
        }
        __syncthreads();

        // ---- LayerNorm z [128], in place ----
        float zs = 0.f, zq = 0.f;
        for (int i = 0; i < 64; ++i) { const float v = zsh[m * 128 + g * 64 + i]; zs += v; zq += v * v; }
        zs = pair_red(zs); zq = pair_red(zq);
        const float zm = zs * (1.f / 128.f);
        const float zr = rsqrtf(zq * (1.f / 128.f) - zm * zm + EPS);
        for (int i = 0; i < 64; ++i) {
            const int c = g * 64 + i;
            zsh[m * 128 + c] = (zsh[m * 128 + c] - zm) * zr * zg[c] + zb[c];
        }
        __syncthreads();
    }

    // ---- affect latent (tanh(av @ aff_w + b)) stored in xsh (x no longer needed) ----
    {
        const float a0 = affect_vec[(row0 + m) * 3 + 0];
        const float a1 = affect_vec[(row0 + m) * 3 + 1];
        const float a2 = affect_vec[(row0 + m) * 3 + 2];
        for (int i = 0; i < 64; ++i) {
            const int c = g * 64 + i;
            xsh[m * 128 + c] = tanhf(a0 * aff_w[c] + a1 * aff_w[128 + c] + a2 * aff_w[256 + c] + aff_b[c]);
        }
        __syncthreads();
        float dp = 0.f;
        for (int i = 0; i < 64; ++i) {
            const int c = g * 64 + i;
            dp += zsh[m * 128 + c] * gate_w[c] + xsh[m * 128 + c] * gate_w[128 + c];
        }
        dp = pair_red(dp);
        const float gate = 1.f / (1.f + expf(-(dp + gate_b[0])));
        for (int i = 0; i < 64; ++i) {
            const int c = g * 64 + i;
            zsh[m * 128 + c] += 0.3f * gate * xsh[m * 128 + c];
        }
        float gs = (g == 0) ? gate : 0.f;
        gs += __shfl_xor(gs, 1, 32); gs += __shfl_xor(gs, 2, 32);
        gs += __shfl_xor(gs, 4, 32); gs += __shfl_xor(gs, 8, 32);
        gs += __shfl_xor(gs, 16, 32);
        if (l == 0) atomicAdd(conf_sum, gs);
        __syncthreads();
    }

    for (int i = 0; i < 64; ++i) {
        const int e = l + 32 * i;
        zbuf[base + e]   = zsh[e];
        hist_k[base + e] = (_Float16)zsh[e];
    }
}

// ---------------------------------------------------------------------------
// K2: attention over history (only the last query matters), z = LN(z + out@o_w)
// ---------------------------------------------------------------------------
__global__ __launch_bounds__(32, 4) void k2_attn(
    float* __restrict__ zbuf, const _Float16* __restrict__ hist, int t,
    const _Float16* __restrict__ qp, const float* __restrict__ qb,
    const _Float16* __restrict__ kp, const float* __restrict__ kb,
    const _Float16* __restrict__ vp, const float* __restrict__ vb,
    const _Float16* __restrict__ op, const float* __restrict__ ob,
    const float* __restrict__ zg, const float* __restrict__ zb) {
    __shared__ _Float16 seq[5 * 16 * 128];
    __shared__ _Float16 kk16[5 * 16 * 128];
    __shared__ _Float16 vv16[5 * 16 * 128];
    __shared__ _Float16 q16[16 * 128];
    __shared__ float    attnout[16 * 128];
    __shared__ float    zsh[16 * 128];

    const int    l = threadIdx.x, m = l & 15, g = l >> 4;
    const size_t base = (size_t)blockIdx.x * 16 * 128;

    for (int s = 0; s < t; ++s)
        for (int i = 0; i < 64; ++i) {
            const int e = l + 32 * i;
            seq[s * 2048 + e] = hist[(size_t)s * BD + base + e];
        }
    for (int i = 0; i < 64; ++i) { const int e = l + 32 * i; zsh[e] = zbuf[base + e]; }
    __syncthreads();

    // projections: K,V for every step; Q only for the last step
#pragma unroll 1
    for (int s = 0; s < t; ++s) {
#pragma unroll 1
        for (int nt0 = 0; nt0 < 8; nt0 += 4) {
            v8f accs[4];
            zero_accs4(accs);
            gemm_block<4, 4>(&seq[s * 2048], 128, kp, nt0, 8, l, accs);
#pragma unroll
            for (int q = 0; q < 4; ++q)
#pragma unroll
                for (int j = 0; j < 8; ++j) {
                    const int mm = j + 8 * g, n = (nt0 + q) * 16 + m;
                    kk16[s * 2048 + mm * 128 + n] = (_Float16)(accs[q][j] + kb[n]);
                }
            zero_accs4(accs);
            gemm_block<4, 4>(&seq[s * 2048], 128, vp, nt0, 8, l, accs);
#pragma unroll
            for (int q = 0; q < 4; ++q)
#pragma unroll
                for (int j = 0; j < 8; ++j) {
                    const int mm = j + 8 * g, n = (nt0 + q) * 16 + m;
                    vv16[s * 2048 + mm * 128 + n] = (_Float16)(accs[q][j] + vb[n]);
                }
            if (s == t - 1) {
                zero_accs4(accs);
                gemm_block<4, 4>(&seq[s * 2048], 128, qp, nt0, 8, l, accs);
#pragma unroll
                for (int q = 0; q < 4; ++q)
#pragma unroll
                    for (int j = 0; j < 8; ++j) {
                        const int mm = j + 8 * g, n = (nt0 + q) * 16 + m;
                        q16[mm * 128 + n] = (_Float16)(accs[q][j] + qb[n]);
                    }
            }
        }
    }
    __syncthreads();

    // softmax over t keys for the last query, per row
    float sc[5];
    float mx = -1e30f;
    for (int s = 0; s < t; ++s) {
        float d = 0.f;
        for (int i = 0; i < 64; ++i) {
            const int c = g * 64 + i;
            d += (float)q16[m * 128 + c] * (float)kk16[s * 2048 + m * 128 + c];
        }
        d = pair_red(d) * 0.08838834764831845f;  // 1/sqrt(128)
        sc[s] = d; mx = fmaxf(mx, d);
    }
    float den = 0.f;
    for (int s = 0; s < t; ++s) { sc[s] = expf(sc[s] - mx); den += sc[s]; }
    const float inv = 1.f / den;
    for (int i = 0; i < 64; ++i) {
        const int c = g * 64 + i;
        float o = 0.f;
        for (int s = 0; s < t; ++s) o += sc[s] * (float)vv16[s * 2048 + m * 128 + c];
        attnout[m * 128 + c] = o * inv;
    }
    __syncthreads();
    for (int i = 0; i < 64; ++i) { const int e = l + 32 * i; q16[e] = (_Float16)attnout[e]; }
    __syncthreads();

    // output projection + residual
#pragma unroll 1
    for (int nt0 = 0; nt0 < 8; nt0 += 4) {
        v8f accs[4];
        zero_accs4(accs);
        gemm_block<4, 4>(q16, 128, op, nt0, 8, l, accs);
#pragma unroll
        for (int q = 0; q < 4; ++q)
#pragma unroll
            for (int j = 0; j < 8; ++j) {
                const int mm = j + 8 * g, n = (nt0 + q) * 16 + m;
                zsh[mm * 128 + n] += accs[q][j] + ob[n];
            }
    }
    __syncthreads();

    // LN z
    float zs = 0.f, zq = 0.f;
    for (int i = 0; i < 64; ++i) { const float v = zsh[m * 128 + g * 64 + i]; zs += v; zq += v * v; }
    zs = pair_red(zs); zq = pair_red(zq);
    const float zm = zs * (1.f / 128.f);
    const float zr = rsqrtf(zq * (1.f / 128.f) - zm * zm + EPS);
    for (int i = 0; i < 64; ++i) {
        const int c = g * 64 + i;
        zsh[m * 128 + c] = (zsh[m * 128 + c] - zm) * zr * zg[c] + zb[c];
    }
    __syncthreads();
    for (int i = 0; i < 64; ++i) { const int e = l + 32 * i; zbuf[base + e] = zsh[e]; }
}

// ---------------------------------------------------------------------------
// scal layout: [0]=conf_sum acc  [1]=sx acc  [2]=sy acc  [3]=faith  [4]=bond
//              [5..9]=conf[k]
// ---------------------------------------------------------------------------
__global__ void kred1(float* __restrict__ scal, const float* __restrict__ memv,
                      float* __restrict__ future, int k) {
    const int tid = threadIdx.x;
    if (tid == 0) {
        const float c = scal[0] * (1.f / (float)BATCH);
        scal[5 + k] = c;
        scal[0] = 0.f;
        float s = 0.f;
        for (int i = 0; i <= k; ++i) s += scal[5 + i];
        scal[3] = sqrtf(s / (float)(k + 1));
    }
    if (tid < 128) {
        const int cnt = (k < 3) ? k : 3;
        float f = 0.f;
        for (int i = 0; i < cnt; ++i) f += memv[(k - 1 - i) * 128 + tid];
        future[tid] = cnt ? f / (float)cnt : 0.f;
    }
}

__global__ void k3_soul(const float* __restrict__ x, const float* __restrict__ ybuf,
                        float* __restrict__ zbuf, const float* __restrict__ faith_p,
                        const float* __restrict__ future, float* __restrict__ sx,
                        float* __restrict__ sy, float* __restrict__ memory_k) {
    __shared__ float rx[256], ry[256];
    const float faith = *faith_p;
    float lx = 0.f, ly = 0.f;
    const size_t stride = (size_t)gridDim.x * blockDim.x;
    for (size_t idx = (size_t)blockIdx.x * blockDim.x + threadIdx.x; idx < BD; idx += stride) {
        float z = zbuf[idx];
        const float y = ybuf[idx], xx = x[idx];
        const float rinse = tanhf(z + y);
        z = z + faith * 0.05f * z;
        lx += xx * z; ly += y * rinse;
        zbuf[idx] = z;
        if (idx < 128) memory_k[idx] = tanhf(rinse + 0.3f * future[idx]);
    }
    rx[threadIdx.x] = lx; ry[threadIdx.x] = ly;
    __syncthreads();
    for (int s = blockDim.x / 2; s > 0; s >>= 1) {
        if ((int)threadIdx.x < s) { rx[threadIdx.x] += rx[threadIdx.x + s]; ry[threadIdx.x] += ry[threadIdx.x + s]; }
        __syncthreads();
    }
    if (threadIdx.x == 0) { atomicAdd(sx, rx[0]); atomicAdd(sy, ry[0]); }
}

__global__ void kred2(float* __restrict__ scal) {
    if (threadIdx.x == 0) {
        const float invn = 1.f / ((float)BATCH * 128.f);
        scal[4] = tanhf(scal[1] * invn + scal[2] * invn);
        scal[1] = 0.f; scal[2] = 0.f;
    }
}

// ---------------------------------------------------------------------------
// K4: bond blend + answer MLP, y += 0.4 * MLP(LN(concat(y,z)))
// ---------------------------------------------------------------------------
__global__ __launch_bounds__(32, 4) void k4_answer(
    float* __restrict__ ybuf, float* __restrict__ zbuf, const float* __restrict__ scal,
    const float* __restrict__ future,
    const float* __restrict__ ag, const float* __restrict__ ab,
    const _Float16* __restrict__ w1p, const float* __restrict__ b1,
    const _Float16* __restrict__ w2p, const float* __restrict__ b2) {
    __shared__ float    ysh[16 * 128], zsh[16 * 128];
    __shared__ _Float16 a16[16 * 256];
    __shared__ _Float16 h16[16 * 256];

    const int    l = threadIdx.x, m = l & 15, g = l >> 4;
    const size_t base = (size_t)blockIdx.x * 16 * 128;
    const float  bond = scal[4];

    for (int i = 0; i < 64; ++i) {
        const int e = l + 32 * i, c = e & 127;
        ysh[e] = ybuf[base + e];
        float z = zbuf[base + e];
        z = z + 0.2f * bond * (future[c] - z);
        zsh[e] = z;
    }
    __syncthreads();

    // LN over concat(y,z) [256]
    float s = 0.f, sq = 0.f;
    for (int i = 0; i < 128; ++i) {
        const int c = g * 128 + i;
        const float v = (c < 128) ? ysh[m * 128 + c] : zsh[m * 128 + c - 128];
        s += v; sq += v * v;
    }
    s = pair_red(s); sq = pair_red(sq);
    const float mean = s * (1.f / 256.f);
    const float rstd = rsqrtf(sq * (1.f / 256.f) - mean * mean + EPS);
    for (int i = 0; i < 128; ++i) {
        const int c = g * 128 + i;
        const float v = (c < 128) ? ysh[m * 128 + c] : zsh[m * 128 + c - 128];
        a16[m * 256 + c] = (_Float16)((v - mean) * rstd * ag[c] + ab[c]);
    }
    __syncthreads();

#pragma unroll 1
    for (int nt0 = 0; nt0 < 16; nt0 += 4) {
        v8f accs[4];
        zero_accs4(accs);
        gemm_block<8, 4>(a16, 256, w1p, nt0, 16, l, accs);
#pragma unroll
        for (int q = 0; q < 4; ++q)
#pragma unroll
            for (int j = 0; j < 8; ++j) {
                const int mm = j + 8 * g, n = (nt0 + q) * 16 + m;
                h16[mm * 256 + n] = (_Float16)gelu_erf(accs[q][j] + b1[n]);
            }
    }
    __syncthreads();

#pragma unroll 1
    for (int nt0 = 0; nt0 < 8; nt0 += 4) {
        v8f accs[4];
        zero_accs4(accs);
        gemm_block<8, 4>(h16, 256, w2p, nt0, 8, l, accs);
#pragma unroll
        for (int q = 0; q < 4; ++q)
#pragma unroll
            for (int j = 0; j < 8; ++j) {
                const int mm = j + 8 * g, n = (nt0 + q) * 16 + m;
                ysh[mm * 128 + n] += 0.4f * (accs[q][j] + b2[n]);
            }
    }
    __syncthreads();
    for (int i = 0; i < 64; ++i) {
        const int e = l + 32 * i;
        ybuf[base + e] = ysh[e];
        zbuf[base + e] = zsh[e];
    }
}

// ---------------------------------------------------------------------------
// Final: pad head + write outputs (y | confs[5] | pad)
// ---------------------------------------------------------------------------
__global__ __launch_bounds__(32, 4) void kfinal(
    const float* __restrict__ ybuf, const float* __restrict__ zbuf,
    const float* __restrict__ scal,
    const _Float16* __restrict__ p1p, const float* __restrict__ pb1,
    const float* __restrict__ pad_w2, const float* __restrict__ pb2,
    float* __restrict__ out) {
    __shared__ _Float16 z16[16 * 128];
    __shared__ _Float16 h16[16 * 64];
    const int    l = threadIdx.x, m = l & 15, g = l >> 4;
    const size_t row0 = (size_t)blockIdx.x * 16;
    const size_t base = row0 * 128;

    for (int i = 0; i < 64; ++i) {
        const int e = l + 32 * i;
        z16[e]        = (_Float16)zbuf[base + e];
        out[base + e] = ybuf[base + e];   // y output
    }
    __syncthreads();

    {
        v8f accs[4];
        zero_accs4(accs);
        gemm_block<4, 4>(z16, 128, p1p, 0, 4, l, accs);
#pragma unroll
        for (int q = 0; q < 4; ++q)
#pragma unroll
            for (int j = 0; j < 8; ++j) {
                const int mm = j + 8 * g, n = q * 16 + m;
                h16[mm * 64 + n] = (_Float16)gelu_erf(accs[q][j] + pb1[n]);
            }
    }
    __syncthreads();

    const size_t padBase = BD + 5;
#pragma unroll
    for (int j3 = 0; j3 < 3; ++j3) {
        float d = 0.f;
        for (int i = 0; i < 32; ++i) {
            const int c = g * 32 + i;
            d += (float)h16[m * 64 + c] * pad_w2[c * 3 + j3];
        }
        d = pair_red(d);
        if (g == 0) out[padBase + (row0 + m) * 3 + j3] = tanhf(d + pb2[j3]);
    }
    if (blockIdx.x == 0 && l < 5) out[BD + l] = scal[5 + l];
}

// ---------------------------------------------------------------------------
extern "C" void kernel_launch(void* const* d_in, const int* in_sizes, int n_in,
                              void* d_out, int out_size, void* d_ws, size_t ws_size,
                              hipStream_t stream) {
    (void)in_sizes; (void)n_in; (void)out_size; (void)ws_size;
    const float* x        = (const float*)d_in[0];
    const float* y_init   = (const float*)d_in[1];
    const float* affect   = (const float*)d_in[2];
    const float* ln_lat_g = (const float*)d_in[3];
    const float* ln_lat_b = (const float*)d_in[4];
    const float* ln_ans_g = (const float*)d_in[5];
    const float* ln_ans_b = (const float*)d_in[6];
    const float* ln_z_g   = (const float*)d_in[7];
    const float* ln_z_b   = (const float*)d_in[8];
    const float* lat_w1   = (const float*)d_in[9];
    const float* lat_b1   = (const float*)d_in[10];
    const float* lat_w2   = (const float*)d_in[11];
    const float* lat_b2   = (const float*)d_in[12];
    const float* ans_w1   = (const float*)d_in[13];
    const float* ans_b1   = (const float*)d_in[14];
    const float* ans_w2   = (const float*)d_in[15];
    const float* ans_b2   = (const float*)d_in[16];
    const float* aff_w    = (const float*)d_in[17];
    const float* aff_b    = (const float*)d_in[18];
    const float* gate_w   = (const float*)d_in[19];
    const float* gate_b   = (const float*)d_in[20];
    const float* q_w      = (const float*)d_in[21];
    const float* q_b      = (const float*)d_in[22];
    const float* k_w      = (const float*)d_in[23];
    const float* k_b      = (const float*)d_in[24];
    const float* v_w      = (const float*)d_in[25];
    const float* v_b      = (const float*)d_in[26];
    const float* o_w      = (const float*)d_in[27];
    const float* o_b      = (const float*)d_in[28];
    const float* pad_w1   = (const float*)d_in[29];
    const float* pad_b1   = (const float*)d_in[30];
    const float* pad_w2   = (const float*)d_in[31];
    const float* pad_b2   = (const float*)d_in[32];
    float* out = (float*)d_out;

    // ---- workspace layout (~152 MB) ----
    float*    zbuf = (float*)d_ws;
    float*    ybuf = zbuf + BD;
    _Float16* hist = (_Float16*)(ybuf + BD);              // 5 * BD halves
    float*    memv = (float*)(hist + (size_t)KOUT * BD);  // 5 * 128
    float*    future = memv + KOUT * 128;                 // 128
    float*    scal   = future + 128;                      // 16
    _Float16* p = (_Float16*)(scal + 16);                 // packed weights (32B aligned)
    _Float16* lat_w1p = p; p += 384 * 256;
    _Float16* lat_w2p = p; p += 256 * 128;
    _Float16* ans_w1p = p; p += 256 * 256;
    _Float16* ans_w2p = p; p += 256 * 128;
    _Float16* qp = p; p += 128 * 128;
    _Float16* kp = p; p += 128 * 128;
    _Float16* vp = p; p += 128 * 128;
    _Float16* op = p; p += 128 * 128;
    _Float16* p1p = p; p += 128 * 64;

    auto pack = [&](const float* W, _Float16* P, int Kd, int Nd) {
        const int tot = Kd * Nd;
        pack_b_kernel<<<(tot + 255) / 256, 256, 0, stream>>>(W, P, Kd, Nd);
    };
    pack(lat_w1, lat_w1p, 384, 256);
    pack(lat_w2, lat_w2p, 256, 128);
    pack(ans_w1, ans_w1p, 256, 256);
    pack(ans_w2, ans_w2p, 256, 128);
    pack(q_w, qp, 128, 128);
    pack(k_w, kp, 128, 128);
    pack(v_w, vp, 128, 128);
    pack(o_w, op, 128, 128);
    pack(pad_w1, p1p, 128, 64);

    kinit<<<4096, 256, 0, stream>>>(zbuf, ybuf, y_init, scal);

    const int tiles = BATCH / 16;  // 4096 wave-tiles
    for (int k = 0; k < KOUT; ++k) {
        k1_inner<<<tiles, 32, 0, stream>>>(
            x, ybuf, zbuf, hist + (size_t)k * BD,
            ln_lat_g, ln_lat_b, ln_z_g, ln_z_b,
            lat_w1p, lat_b1, lat_w2p, lat_b2,
            affect, aff_w, aff_b, gate_w, gate_b, scal /* conf_sum */);
        if (k > 0)
            k2_attn<<<tiles, 32, 0, stream>>>(
                zbuf, hist, k + 1, qp, q_b, kp, k_b, vp, v_b, op, o_b, ln_z_g, ln_z_b);
        kred1<<<1, 128, 0, stream>>>(scal, memv, future, k);
        k3_soul<<<2048, 256, 0, stream>>>(x, ybuf, zbuf, scal + 3, future,
                                          scal + 1, scal + 2, memv + k * 128);
        kred2<<<1, 32, 0, stream>>>(scal);
        k4_answer<<<tiles, 32, 0, stream>>>(ybuf, zbuf, scal, future,
                                            ln_ans_g, ln_ans_b,
                                            ans_w1p, ans_b1, ans_w2p, ans_b2);
    }
    kfinal<<<tiles, 32, 0, stream>>>(ybuf, zbuf, scal, p1p, pad_b1, pad_w2, pad_b2, out);
}
